// FusionSwinTransformerBlock_16844861735416
// MI455X (gfx1250) — matrix operand order repaired and checked by reading, not compiled, not run on hardware
//
#include <hip/hip_runtime.h>
#include <math.h>

// ---------------- constants ----------------
#define CDIM   180          // channels
#define CPAD   192          // channels padded to multiple of 32
#define TOKS   65536        // B*H*W = 4*128*128 tokens (== windows*64)
#define NWINS  1024         // total windows (4 * 256)
#define NHEADS 6
#define HDIM   30
#define HPAD   32
#define MLPH   720
#define MLPHP  736
#define C2P    384          // 2C=360 padded

// ---------------- WMMA fragment types ----------------
typedef __attribute__((ext_vector_type(16))) _Float16     v16h;
typedef __attribute__((ext_vector_type(8)))  float        v8f;
typedef __attribute__((ext_vector_type(4)))  unsigned int v4u;

union H16 { v16h v; v4u q[2]; _Float16 h[16]; };

// A-matrix 16x32 f16 layout (ISA 7.12.2): lane group g=lane>>4,
// elements 0..7 = K[8g .. 8g+7], elements 8..15 = K[16+8g .. 23+8g]
__device__ __forceinline__ v16h ldfrag_a(const _Float16* row, int g) {
  H16 u;
  u.q[0] = *(const v4u*)(row + 8 * g);
  u.q[1] = *(const v4u*)(row + 16 + 8 * g);
  return u.v;
}
// B-matrix 32x16 f16: lane n holds K[8g .. 8g+7], K[16+8g .. 23+8g] (same order as A)
__device__ __forceinline__ v16h ldfrag_b(const _Float16* row, int g) {
  H16 u;
  u.q[0] = *(const v4u*)(row + 8 * g);
  u.q[1] = *(const v4u*)(row + 16 + 8 * g);
  return u.v;
}
__device__ __forceinline__ v8f wmma16(v16h a, v16h b, v8f c) {
  return __builtin_amdgcn_wmma_f32_16x16x32_f16(false, a, false, b,
                                                (short)0, c, false, false);
}
__device__ __forceinline__ float gelu_f(float x) {
  return 0.5f * x * (1.f + erff(x * 0.70710678118654752f));
}
// token (window order) -> original (b,h,w); identical map used for LN1 read,
// residual shortcut read, and final scatter (roll cancels).
__device__ __forceinline__ void decode_tok(int t, int& b, int& h, int& w) {
  int win = t >> 6, pos = t & 63;
  b = win >> 8;
  int widx = win & 255;
  h = (((widx >> 4) << 3) + (pos >> 3) + 4) & 127;
  w = (((widx & 15) << 3) + (pos & 7) + 4) & 127;
}
__device__ __forceinline__ int sregion(int x) { return x < 120 ? 0 : (x < 124 ? 1 : 2); }

// ---------------- weight conversion: f32 [N,K] -> f16 [N,Kp] zero-padded ----------------
__global__ void convert_w(const float* __restrict__ W, _Float16* __restrict__ Wh,
                          int N, int K, int Kp) {
  int i = blockIdx.x * blockDim.x + threadIdx.x;
  if (i >= N * Kp) return;
  int n = i / Kp, k = i % Kp;
  Wh[i] = (k < K) ? (_Float16)W[(size_t)n * K + k] : (_Float16)0.f;
}

// ---------------- LN1 + shift + window partition -> f16 padded ----------------
__global__ void ln_shift_win(const float* __restrict__ x, const float* __restrict__ gamma,
                             const float* __restrict__ beta, _Float16* __restrict__ XW) {
  int t = blockIdx.x * blockDim.x + threadIdx.x;
  if (t >= TOKS) return;
  int b, h, w; decode_tok(t, b, h, w);
  const float* src = x + (size_t)b * CDIM * 16384 + (size_t)h * 128 + w;
  float s = 0.f, ss = 0.f;
  for (int c = 0; c < CDIM; ++c) { float v = src[(size_t)c * 16384]; s += v; ss += v * v; }
  float mu = s * (1.f / CDIM);
  float rstd = rsqrtf(ss * (1.f / CDIM) - mu * mu + 1e-5f);
  _Float16* dst = XW + (size_t)t * CPAD;
  for (int c = 0; c < CDIM; ++c) {
    float v = src[(size_t)c * 16384];
    dst[c] = (_Float16)((v - mu) * rstd * gamma[c] + beta[c]);
  }
  for (int c = CDIM; c < CPAD; ++c) dst[c] = (_Float16)0.f;
}

// ---------------- generic WMMA GEMM: D = A[M,Kp] x Wh[Nt,Kp]^T + bias ----------------
// 4 waves / block, each wave computes a 32x32 tile (2x2 register blocking:
// 4 WMMA per 8 b128 loads -> 2x the arithmetic intensity of 1-tile waves).
// Block macro-tile = 64x64.
// mode 0: f32 out [M,ldo], act 0 none / 1 gelu / 2 sigmoid
// mode 1: f16 out [M,ldo], gelu if act==1
// mode 2: QKV scatter into per-(window,head) [64,HPAD] f16 blocks (q scaled)
__global__ __launch_bounds__(128) void gemm_wmma(
    const _Float16* __restrict__ A, const _Float16* __restrict__ W,
    const float* __restrict__ bias, int Nt, int Kp, int mode, int act,
    float* __restrict__ outF, _Float16* __restrict__ outH, int ldo, float qscale,
    _Float16* __restrict__ qout, _Float16* __restrict__ kout, _Float16* __restrict__ vout) {
  const int wave = threadIdx.x >> 5;
  const int lane = threadIdx.x & 31;
  const int g = lane >> 4;
  const int ln = lane & 15;
  const int wm = wave >> 1, wn = wave & 1;
  const int m0 = blockIdx.y * 64 + wm * 32;    // rows m0 .. m0+31
  const int n0 = blockIdx.x * 64 + wn * 32;    // cols n0 .. n0+31
  const int nc0 = n0 + ln;                     // this lane's col in n-subtile 0
  const int nc1 = n0 + 16 + ln;                // ... in n-subtile 1
  const bool v0 = (nc0 < Nt), v1 = (nc1 < Nt);
  const _Float16* arow0 = A + (size_t)(m0 + ln) * Kp;
  const _Float16* arow1 = A + (size_t)(m0 + 16 + ln) * Kp;
  const _Float16* wrow0 = W + (size_t)(v0 ? nc0 : 0) * Kp;  // garbage cols masked at store
  const _Float16* wrow1 = W + (size_t)(v1 ? nc1 : 0) * Kp;

  v8f acc[2][2];
  const v8f zz = {0.f, 0.f, 0.f, 0.f, 0.f, 0.f, 0.f, 0.f};
  acc[0][0] = zz; acc[0][1] = zz; acc[1][0] = zz; acc[1][1] = zz;

  for (int k = 0; k < Kp; k += 32) {
    __builtin_prefetch(arow0 + k + 128, 0, 0);             // global_prefetch_b8
    __builtin_prefetch(arow1 + k + 128, 0, 0);
    v16h a0 = ldfrag_a(arow0 + k, g);
    v16h a1 = ldfrag_a(arow1 + k, g);
    v16h b0 = ldfrag_b(wrow0 + k, g);
    v16h b1 = ldfrag_b(wrow1 + k, g);
    acc[0][0] = wmma16(a0, b0, acc[0][0]);
    acc[0][1] = wmma16(a0, b1, acc[0][1]);
    acc[1][0] = wmma16(a1, b0, acc[1][0]);
    acc[1][1] = wmma16(a1, b1, acc[1][1]);
  }

#pragma unroll
  for (int j = 0; j < 2; ++j) {
    const int n = n0 + j * 16 + ln;
    if (n >= Nt) continue;
    const float bv = bias[n];
    if (mode == 2) {
      const int which = n / CDIM, rem = n % CDIM, head = rem / HDIM, dim = rem % HDIM;
#pragma unroll
      for (int i = 0; i < 2; ++i) {
#pragma unroll
        for (int r = 0; r < 8; ++r) {
          const int m = m0 + i * 16 + r + 8 * g;
          const size_t off = (((size_t)(m >> 6) * NHEADS + head) * 64 + (m & 63)) * HPAD + dim;
          const float v = acc[i][j][r] + bv;
          if (which == 0)      qout[off] = (_Float16)(v * qscale);
          else if (which == 1) kout[off] = (_Float16)v;
          else                 vout[off] = (_Float16)v;
        }
      }
    } else if (mode == 1) {
#pragma unroll
      for (int i = 0; i < 2; ++i) {
#pragma unroll
        for (int r = 0; r < 8; ++r) {
          const int m = m0 + i * 16 + r + 8 * g;
          float v = acc[i][j][r] + bv;
          if (act == 1) v = gelu_f(v);
          outH[(size_t)m * ldo + n] = (_Float16)v;
        }
      }
    } else {
#pragma unroll
      for (int i = 0; i < 2; ++i) {
#pragma unroll
        for (int r = 0; r < 8; ++r) {
          const int m = m0 + i * 16 + r + 8 * g;
          float v = acc[i][j][r] + bv;
          if (act == 1) v = gelu_f(v);
          else if (act == 2) v = 1.f / (1.f + __expf(-v));
          outF[(size_t)m * ldo + n] = v;
        }
      }
    }
  }
}

// ---------------- fused window attention: one block per (window, head) ----------------
__global__ __launch_bounds__(128) void attn_wmma(
    const _Float16* __restrict__ Q, const _Float16* __restrict__ K,
    const _Float16* __restrict__ V, const float* __restrict__ rpb_table,
    float* __restrict__ outAtt) {
  __shared__ __align__(16) float    S[64 * 64];     // 16KB scores, later overlaid by Vt
  __shared__ __align__(16) _Float16 Pm[64 * 64];    // 8KB probabilities (f16)
  _Float16* Vt = (_Float16*)S;                      // 32x64 transposed V overlay

  const int blk = blockIdx.x;              // win*NHEADS + head
  const int win = blk / NHEADS, head = blk % NHEADS;
  const int widx = win & 255;
  const int wh = widx >> 4, ww = widx & 15;
  const _Float16* Qh = Q + (size_t)blk * 64 * HPAD;
  const _Float16* Kh = K + (size_t)blk * 64 * HPAD;
  const _Float16* Vh = V + (size_t)blk * 64 * HPAD;

  const int wave = threadIdx.x >> 5;
  const int lane = threadIdx.x & 31;
  const int g = lane >> 4;
  const int ln = lane & 15;

  // ---- S = Q * K^T (+rpb +mask): wave handles row tile mt=wave, all 4 col tiles
  const int mt = wave;
  v16h af = ldfrag_a(Qh + (size_t)(mt * 16 + ln) * HPAD, g);
  for (int nt = 0; nt < 4; ++nt) {
    v8f acc = {0.f, 0.f, 0.f, 0.f, 0.f, 0.f, 0.f, 0.f};
    v16h bf = ldfrag_b(Kh + (size_t)(nt * 16 + ln) * HPAD, g);
    acc = wmma16(af, bf, acc);
    int col = nt * 16 + ln;                 // key position
    int mh = col >> 3, mw = col & 7;
    int gk = sregion(wh * 8 + mh) * 3 + sregion(ww * 8 + mw);
    for (int r = 0; r < 8; ++r) {
      int row = mt * 16 + r + 8 * g;        // query position
      int nh = row >> 3, nw = row & 7;
      int idx = (nh - mh + 7) * 15 + (nw - mw + 7);
      float rv = rpb_table[idx * NHEADS + head];
      int gq = sregion(wh * 8 + nh) * 3 + sregion(ww * 8 + nw);
      float mv = (gq != gk) ? -100.f : 0.f;
      S[row * 64 + col] = acc[r] + rv + mv;
    }
  }
  __syncthreads();

  // ---- softmax over keys, write P (f16)
  if (threadIdx.x < 64) {
    int row = threadIdx.x;
    float mx = -1e30f;
    for (int m = 0; m < 64; ++m) mx = fmaxf(mx, S[row * 64 + m]);
    float sum = 0.f;
    for (int m = 0; m < 64; ++m) {
      float e = __expf(S[row * 64 + m] - mx);
      S[row * 64 + m] = e;
      sum += e;
    }
    float inv = 1.f / sum;
    for (int m = 0; m < 64; ++m) Pm[row * 64 + m] = (_Float16)(S[row * 64 + m] * inv);
  }
  __syncthreads();

  // ---- transpose V into LDS (Vt[dim][tok]) over the S region
  for (int i = threadIdx.x; i < 64 * HPAD; i += 128) {
    int tok = i >> 5, dim = i & 31;
    Vt[dim * 64 + tok] = Vh[tok * HPAD + dim];
  }
  __syncthreads();

  // ---- O = P * V : wave -> row tile mt, col tiles nt=0..1, K=64 in two steps
  for (int nt = 0; nt < 2; ++nt) {
    v8f acc = {0.f, 0.f, 0.f, 0.f, 0.f, 0.f, 0.f, 0.f};
    for (int ks = 0; ks < 64; ks += 32) {
      v16h pa = ldfrag_a(Pm + (size_t)(mt * 16 + ln) * 64 + ks, g);
      v16h vb = ldfrag_b(Vt + (size_t)(nt * 16 + ln) * 64 + ks, g);
      acc = wmma16(pa, vb, acc);
    }
    int n = nt * 16 + ln;
    if (n < HDIM) {
      for (int r = 0; r < 8; ++r) {
        int row = mt * 16 + r + 8 * g;
        outAtt[((size_t)win * 64 + row) * CDIM + head * HDIM + n] = acc[r];
      }
    }
  }
}

// ---------------- gating + concat -> f16 proj input [TOKS, C2P] ----------------
__global__ void build_aproj(const float* __restrict__ ATTe, const float* __restrict__ ATTb,
                            const float* __restrict__ SIGe, const float* __restrict__ SIGb,
                            _Float16* __restrict__ APe, _Float16* __restrict__ APb) {
  int t = blockIdx.x * blockDim.x + threadIdx.x;
  if (t >= TOKS) return;
  size_t base = (size_t)t * CDIM;
  size_t ob = (size_t)t * C2P;
  for (int c = 0; c < CDIM; ++c) {
    float ae = ATTe[base + c], ab = ATTb[base + c];
    float se = SIGe[base + c], sb = SIGb[base + c];
    APe[ob + c] = (_Float16)(ae * se);
    APe[ob + CDIM + c] = (_Float16)(ab * se);
    APb[ob + c] = (_Float16)(ae * sb);
    APb[ob + CDIM + c] = (_Float16)(ab * sb);
  }
  for (int c = 2 * CDIM; c < C2P; ++c) { APe[ob + c] = (_Float16)0.f; APb[ob + c] = (_Float16)0.f; }
}

// ---------------- residual1 + LN2 -> X1 (f32) and XH (f16 padded) ----------------
__global__ void res1_ln2(const float* __restrict__ x_in, const float* __restrict__ O,
                         const float* __restrict__ g2, const float* __restrict__ b2,
                         float* __restrict__ X1, _Float16* __restrict__ XH) {
  int t = blockIdx.x * blockDim.x + threadIdx.x;
  if (t >= TOKS) return;
  int b, h, w; decode_tok(t, b, h, w);
  const float* src = x_in + (size_t)b * CDIM * 16384 + (size_t)h * 128 + w;
  const float* o = O + (size_t)t * CDIM;
  float* x1 = X1 + (size_t)t * CDIM;
  float s = 0.f, ss = 0.f;
  for (int c = 0; c < CDIM; ++c) {
    float v = src[(size_t)c * 16384] + o[c];
    x1[c] = v; s += v; ss += v * v;
  }
  float mu = s * (1.f / CDIM);
  float rstd = rsqrtf(ss * (1.f / CDIM) - mu * mu + 1e-5f);
  _Float16* dst = XH + (size_t)t * CPAD;
  for (int c = 0; c < CDIM; ++c)
    dst[c] = (_Float16)((x1[c] - mu) * rstd * g2[c] + b2[c]);
  for (int c = CDIM; c < CPAD; ++c) dst[c] = (_Float16)0.f;
}

// ---------------- final residual + scatter to (B,C,H,W) ----------------
__global__ void final_out(const float* __restrict__ X1, const float* __restrict__ M2,
                          float* __restrict__ out) {
  int t = blockIdx.x * blockDim.x + threadIdx.x;
  if (t >= TOKS) return;
  int b, h, w; decode_tok(t, b, h, w);
  float* dst = out + (size_t)b * CDIM * 16384 + (size_t)h * 128 + w;
  const float* x1 = X1 + (size_t)t * CDIM;
  const float* m2 = M2 + (size_t)t * CDIM;
  for (int c = 0; c < CDIM; ++c) dst[(size_t)c * 16384] = x1[c] + m2[c];
}

// ---------------- host ----------------
extern "C" void kernel_launch(void* const* d_in, const int* in_sizes, int n_in,
                              void* d_out, int out_size, void* d_ws, size_t ws_size,
                              hipStream_t stream) {
  const float* e_f = (const float*)d_in[0];
  const float* b_f = (const float*)d_in[1];
  const float* n1eg = (const float*)d_in[2];  const float* n1eb = (const float*)d_in[3];
  const float* n1bg = (const float*)d_in[4];  const float* n1bb = (const float*)d_in[5];
  const float* qkvew = (const float*)d_in[6]; const float* qkveb = (const float*)d_in[7];
  const float* qkvbw = (const float*)d_in[8]; const float* qkvbb = (const float*)d_in[9];
  const float* psew = (const float*)d_in[10]; const float* pseb = (const float*)d_in[11];
  const float* psbw = (const float*)d_in[12]; const float* psbb = (const float*)d_in[13];
  const float* rpbt = (const float*)d_in[14];
  const float* prew = (const float*)d_in[15]; const float* preb = (const float*)d_in[16];
  const float* prbw = (const float*)d_in[17]; const float* prbb = (const float*)d_in[18];
  const float* n2eg = (const float*)d_in[19]; const float* n2eb = (const float*)d_in[20];
  const float* n2bg = (const float*)d_in[21]; const float* n2bb = (const float*)d_in[22];
  const float* m1ew = (const float*)d_in[23]; const float* m1eb = (const float*)d_in[24];
  const float* m2ew = (const float*)d_in[25]; const float* m2eb = (const float*)d_in[26];
  const float* m1bw = (const float*)d_in[27]; const float* m1bb = (const float*)d_in[28];
  const float* m2bw = (const float*)d_in[29]; const float* m2bb = (const float*)d_in[30];

  char* ws = (char*)d_ws;
  const size_t XWB  = (size_t)TOKS * CPAD * 2;                  // 25,165,824 B
  const size_t QKV1 = (size_t)NWINS * NHEADS * 64 * HPAD * 2;   // 25,165,824 B
  const size_t SIG1 = (size_t)TOKS * CDIM * 4;                  // 47,185,920 B

  size_t off = 0;
  size_t offXWe = off; off += XWB;
  size_t offXWb = off; off += XWB;
  size_t offR1  = off; off += 6 * QKV1;    // QKV -> APROJ -> HH (per-stream)
  size_t offR2  = off; off += 2 * SIG1;    // SIG -> O -> M2
  size_t offR3  = off; off += 2 * SIG1;    // ATT -> X1
  size_t offWgt = off;

  _Float16* XWe = (_Float16*)(ws + offXWe);
  _Float16* XWb = (_Float16*)(ws + offXWb);
  _Float16* XHe = XWe;                          // alias (XW consumed before res1)
  _Float16* XHb = XWb;
  _Float16* Qe = (_Float16*)(ws + offR1);
  _Float16* Ke = (_Float16*)(ws + offR1 + QKV1);
  _Float16* Ve = (_Float16*)(ws + offR1 + 2 * QKV1);
  _Float16* Qb = (_Float16*)(ws + offR1 + 3 * QKV1);
  _Float16* Kb = (_Float16*)(ws + offR1 + 4 * QKV1);
  _Float16* Vb = (_Float16*)(ws + offR1 + 5 * QKV1);
  _Float16* APe = (_Float16*)(ws + offR1);                      // alias over QKV
  _Float16* APb = (_Float16*)(ws + offR1 + (size_t)TOKS * C2P * 2);
  _Float16* HH  = (_Float16*)(ws + offR1);                      // alias, per-stream serial
  float* SIGe = (float*)(ws + offR2);
  float* SIGb = (float*)(ws + offR2 + SIG1);
  float* Oe   = SIGe;                                           // alias (SIG consumed)
  float* Ob   = SIGb;
  float* M2e  = SIGe;                                           // alias (O consumed)
  float* M2b  = SIGb;
  float* ATTe = (float*)(ws + offR3);
  float* ATTb = (float*)(ws + offR3 + SIG1);
  float* X1e  = ATTe;                                           // alias (ATT consumed)
  float* X1b  = ATTb;

  // f16 weights
  _Float16* wp = (_Float16*)(ws + offWgt);
  _Float16* whQKVe = wp; wp += 540 * CPAD;
  _Float16* whQKVb = wp; wp += 540 * CPAD;
  _Float16* whPSe  = wp; wp += CDIM * CPAD;
  _Float16* whPSb  = wp; wp += CDIM * CPAD;
  _Float16* whPRe  = wp; wp += CDIM * C2P;
  _Float16* whPRb  = wp; wp += CDIM * C2P;
  _Float16* whM1e  = wp; wp += MLPH * CPAD;
  _Float16* whM1b  = wp; wp += MLPH * CPAD;
  _Float16* whM2e  = wp; wp += CDIM * MLPHP;
  _Float16* whM2b  = wp; wp += CDIM * MLPHP;

  auto convW = [&](const float* W, _Float16* Wh, int N, int K, int Kp) {
    int total = N * Kp;
    convert_w<<<(total + 255) / 256, 256, 0, stream>>>(W, Wh, N, K, Kp);
  };
  convW(qkvew, whQKVe, 540, CDIM, CPAD);
  convW(qkvbw, whQKVb, 540, CDIM, CPAD);
  convW(psew, whPSe, CDIM, CDIM, CPAD);
  convW(psbw, whPSb, CDIM, CDIM, CPAD);
  convW(prew, whPRe, CDIM, 2 * CDIM, C2P);
  convW(prbw, whPRb, CDIM, 2 * CDIM, C2P);
  convW(m1ew, whM1e, MLPH, CDIM, CPAD);
  convW(m1bw, whM1b, MLPH, CDIM, CPAD);
  convW(m2ew, whM2e, CDIM, MLPH, MLPHP);
  convW(m2bw, whM2b, CDIM, MLPH, MLPHP);

  // zero QKV region so padded head dims (30,31) are exactly 0
  hipMemsetAsync(ws + offR1, 0, 6 * QKV1, stream);

  // LN1 + shift + window partition
  ln_shift_win<<<TOKS / 256, 256, 0, stream>>>(e_f, n1eg, n1eb, XWe);
  ln_shift_win<<<TOKS / 256, 256, 0, stream>>>(b_f, n1bg, n1bb, XWb);

  const float qscale = 0.18257418583505537f;   // 30^-0.5
  auto gemm = [&](const _Float16* A, const _Float16* W, const float* bias,
                  int Nt, int Kp, int mode, int act, float* oF, _Float16* oH, int ldo,
                  _Float16* q, _Float16* k, _Float16* v) {
    dim3 grid((Nt + 63) / 64, TOKS / 64);
    gemm_wmma<<<grid, 128, 0, stream>>>(A, W, bias, Nt, Kp, mode, act,
                                        oF, oH, ldo, qscale, q, k, v);
  };

  // QKV + sigmoid gates
  gemm(XWe, whQKVe, qkveb, 540, CPAD, 2, 0, nullptr, nullptr, 0, Qe, Ke, Ve);
  gemm(XWb, whQKVb, qkvbb, 540, CPAD, 2, 0, nullptr, nullptr, 0, Qb, Kb, Vb);
  gemm(XWe, whPSe, pseb, CDIM, CPAD, 0, 2, SIGe, nullptr, CDIM, nullptr, nullptr, nullptr);
  gemm(XWb, whPSb, psbb, CDIM, CPAD, 0, 2, SIGb, nullptr, CDIM, nullptr, nullptr, nullptr);

  // window attention
  attn_wmma<<<NWINS * NHEADS, 128, 0, stream>>>(Qe, Ke, Ve, rpbt, ATTe);
  attn_wmma<<<NWINS * NHEADS, 128, 0, stream>>>(Qb, Kb, Vb, rpbt, ATTb);

  // gate + concat, then output projections
  build_aproj<<<TOKS / 256, 256, 0, stream>>>(ATTe, ATTb, SIGe, SIGb, APe, APb);
  gemm(APe, whPRe, preb, CDIM, C2P, 0, 0, Oe, nullptr, CDIM, nullptr, nullptr, nullptr);
  gemm(APb, whPRb, prbb, CDIM, C2P, 0, 0, Ob, nullptr, CDIM, nullptr, nullptr, nullptr);

  // residual + LN2
  res1_ln2<<<TOKS / 256, 256, 0, stream>>>(e_f, Oe, n2eg, n2eb, X1e, XHe);
  res1_ln2<<<TOKS / 256, 256, 0, stream>>>(b_f, Ob, n2bg, n2bb, X1b, XHb);

  // MLPs (HH buffer reused serially per stream)
  gemm(XHe, whM1e, m1eb, MLPH, CPAD, 1, 1, nullptr, HH, MLPHP, nullptr, nullptr, nullptr);
  gemm(HH, whM2e, m2eb, CDIM, MLPHP, 0, 0, M2e, nullptr, CDIM, nullptr, nullptr, nullptr);
  gemm(XHb, whM1b, m1bb, MLPH, CPAD, 1, 1, nullptr, HH, MLPHP, nullptr, nullptr, nullptr);
  gemm(HH, whM2b, m2bb, CDIM, MLPHP, 0, 0, M2b, nullptr, CDIM, nullptr, nullptr, nullptr);

  // final residual + scatter to (B,C,H,W); e_out then b_out
  float* out_e = (float*)d_out;
  float* out_b = out_e + (size_t)4 * CDIM * 16384;
  final_out<<<TOKS / 256, 256, 0, stream>>>(X1e, M2e, out_e);
  final_out<<<TOKS / 256, 256, 0, stream>>>(X1b, M2b, out_b);
}
